// MobileMQA_5729486373267
// MI455X (gfx1250) — matrix operand order repaired and checked by reading, not compiled, not run on hardware
//
#include <hip/hip_runtime.h>

// ---------------------------------------------------------------------------
// CDNA5 (gfx1250) wave32 WMMA bf16 implementation of MobileMQA.
// D = A(16x32 bf16) x B(32x16 bf16) + C(16x16 f32) via v_wmma_f32_16x16x32_bf16
// K-chunk / out-proj A-slab staging uses GLOBAL_LOAD_ASYNC_TO_LDS_B128
// (ASYNCcnt) with per-lane padded LDS destinations.
// ---------------------------------------------------------------------------

typedef __attribute__((ext_vector_type(16))) __bf16 v16bf;
typedef __attribute__((ext_vector_type(8)))  float  v8f;

union Frag16 { v16bf v; unsigned int u[8]; };

__device__ __forceinline__ v8f wmma_bf16(const v16bf& a, const v16bf& b, v8f c) {
  return __builtin_amdgcn_wmma_f32_16x16x32_bf16(
      /*neg_a=*/false, a, /*neg_b=*/false, b,
      /*c_mod=*/(short)0, c, /*reuse_a=*/false, /*reuse_b=*/false);
}

// A-operand K index for vgpr-pair j (16-bit A 16x32, contiguous: k = 16*hf + 2*j)
__device__ __forceinline__ int a_koff(int j, int hf) {
  return hf * 16 + j * 2;
}

// Async copy of 16 bytes global -> LDS (per-lane addresses, ASYNCcnt tracked)
__device__ __forceinline__ void async_b128(unsigned lds_off, const void* gptr) {
  unsigned long long ga = (unsigned long long)(uintptr_t)gptr;
  asm volatile("global_load_async_to_lds_b128 %0, %1, off"
               :: "v"(lds_off), "v"(ga) : "memory");
}
__device__ __forceinline__ void wait_async0() {
  asm volatile("s_wait_asynccnt 0" ::: "memory");
}

static constexpr int DIM   = 256;
static constexpr int HD    = 64;
static constexpr int B_    = 8;
static constexpr int M_    = 32768;              // B*64*64
static constexpr int KV    = 1024;               // 32*32

// ===========================================================================
// Kernel 1: fused projections  X[32768,256] x {q_w,k_w,v_w}
// grid (256, 6): y = 0..3 -> q cols, 4 -> k, 5 -> v.  Block tile 128x64.
// (fp32 sources need conversion, so staging stays on the manual path)
// ===========================================================================
__global__ __launch_bounds__(256) void proj_kernel(
    const float* __restrict__ x,
    const float* __restrict__ q_w, const float* __restrict__ q_b,
    const float* __restrict__ k_w, const float* __restrict__ k_b,
    const float* __restrict__ v_w, const float* __restrict__ v_b,
    __bf16* __restrict__ qo,   // [M,256] bf16
    float*  __restrict__ ko,   // [M,64] fp32 (pre-conv)
    float*  __restrict__ vo)   // [M,64] fp32
{
  __shared__ __bf16 As[128][40];   // [m][k], pitch 40
  __shared__ __bf16 Bs[64][40];    // transposed: [n][k], pitch 40

  const int tid  = threadIdx.x;
  const int lane = tid & 31;
  const int wave = tid >> 5;
  const int hf   = lane >> 4;
  const int ln   = lane & 15;
  const int m0   = blockIdx.x * 128;
  const int nb   = blockIdx.y;

  const float* wsrc; int ldw;
  if (nb < 4)       { wsrc = q_w + nb * 64; ldw = DIM; }
  else if (nb == 4) { wsrc = k_w;           ldw = HD;  }
  else              { wsrc = v_w;           ldw = HD;  }

  v8f acc[4];
  #pragma unroll
  for (int t = 0; t < 4; ++t) acc[t] = {0.f,0.f,0.f,0.f,0.f,0.f,0.f,0.f};

  for (int k0 = 0; k0 < DIM; k0 += 32) {
    #pragma unroll
    for (int i = 0; i < 16; ++i) {            // A: 128x32 fp32 -> bf16
      int flat = tid * 16 + i;
      int r = flat >> 5, c = flat & 31;
      As[r][c] = (__bf16)x[(size_t)(m0 + r) * DIM + k0 + c];
    }
    #pragma unroll
    for (int i = 0; i < 8; ++i) {             // B: 32x64, store transposed
      int flat = tid * 8 + i;
      int kk = flat >> 6, n = flat & 63;
      Bs[n][kk] = (__bf16)wsrc[(size_t)(k0 + kk) * ldw + n];
    }
    __syncthreads();

    Frag16 a;
    const int arow = wave * 16 + ln;
    #pragma unroll
    for (int j = 0; j < 8; ++j)
      a.u[j] = *(const unsigned int*)&As[arow][a_koff(j, hf)];

    #pragma unroll
    for (int t = 0; t < 4; ++t) {
      Frag16 bfr;
      const int n = t * 16 + ln;
      #pragma unroll
      for (int j = 0; j < 8; ++j)
        bfr.u[j] = *(const unsigned int*)&Bs[n][hf * 16 + 2 * j];
      acc[t] = wmma_bf16(a.v, bfr.v, acc[t]);
    }
    __syncthreads();
  }

  // epilogue: branch hoisted out of the store loops
  if (nb < 4) {
    const int nc = nb * 64;
    #pragma unroll
    for (int t = 0; t < 4; ++t) {
      const int n = nc + t * 16 + ln;
      const float bv = q_b[n];
      #pragma unroll
      for (int r = 0; r < 8; ++r) {
        const int m = m0 + wave * 16 + r + hf * 8;
        qo[(size_t)m * DIM + n] = (__bf16)(acc[t][r] + bv);
      }
    }
  } else if (nb == 4) {
    #pragma unroll
    for (int t = 0; t < 4; ++t) {
      const int n = t * 16 + ln;
      const float bv = k_b[n];
      #pragma unroll
      for (int r = 0; r < 8; ++r) {
        const int m = m0 + wave * 16 + r + hf * 8;
        ko[(size_t)m * HD + n] = acc[t][r] + bv;
      }
    }
  } else {
    #pragma unroll
    for (int t = 0; t < 4; ++t) {
      const int n = t * 16 + ln;
      const float bv = v_b[n];
      #pragma unroll
      for (int r = 0; r < 8; ++r) {
        const int m = m0 + wave * 16 + r + hf * 8;
        vo[(size_t)m * HD + n] = acc[t][r] + bv;
      }
    }
  }
}

// ===========================================================================
// Kernel 2: depthwise 3x3 stride-2 conv, pad 1.  [B,64,64,64] -> [B,32,32,64]
// grid (2048, 2): y==0 -> k path, y==1 -> v path
// ===========================================================================
__global__ __launch_bounds__(256) void dwconv_kernel(
    const float* __restrict__ kin, const float* __restrict__ kw,
    const float* __restrict__ kb,
    const float* __restrict__ vin, const float* __restrict__ vw,
    const float* __restrict__ vb,
    __bf16* __restrict__ kout, __bf16* __restrict__ vout)
{
  const int idx = blockIdx.x * 256 + threadIdx.x;    // [0, 524288)
  const float* in; const float* w; const float* bia; __bf16* outp;
  if (blockIdx.y == 0) { in = kin; w = kw; bia = kb; outp = kout; }
  else                 { in = vin; w = vw; bia = vb; outp = vout; }

  const int c  = idx & 63;
  const int ow = (idx >> 6) & 31;
  const int oh = (idx >> 11) & 31;
  const int bb = idx >> 16;

  float s = bia[c];
  #pragma unroll
  for (int dy = 0; dy < 3; ++dy) {
    const int ih = oh * 2 - 1 + dy;
    if (ih < 0 || ih >= 64) continue;
    #pragma unroll
    for (int dx = 0; dx < 3; ++dx) {
      const int iw = ow * 2 - 1 + dx;
      if (iw < 0 || iw >= 64) continue;
      s += in[(((size_t)bb * 64 + ih) * 64 + iw) * 64 + c] * w[(dy * 3 + dx) * 64 + c];
    }
  }
  outp[idx] = (__bf16)s;
}

// ===========================================================================
// Kernel 3: flash MQA attention.
// grid 1024 blocks; block = 8 waves = 4 heads x 2 q-tiles over 32 q rows.
// K chunk (contiguous 4KB bf16) async-copied into padded LDS rows.
// ===========================================================================
__global__ __launch_bounds__(256) void attn_kernel(
    const __bf16* __restrict__ q,    // [M,256]
    const __bf16* __restrict__ kc,   // [B*1024,64]
    const __bf16* __restrict__ vc,   // [B*1024,64]
    __bf16* __restrict__ ao)         // [M,256]
{
  __shared__ __bf16 Ks[32][72];       // [key][c], pitch 144B (9x16B)
  __shared__ __bf16 Vt[64][40];       // [d][key]  (PV B-operand: pairs in key)
  __shared__ __bf16 Ps[8][16][40];    // per-wave P staging [qrow][key]

  const int tid   = threadIdx.x;
  const int lane  = tid & 31;
  const int wave  = tid >> 5;
  const int hf    = lane >> 4;
  const int ln    = lane & 15;
  const int head  = wave >> 1;
  const int qbase = blockIdx.x * 32 + (wave & 1) * 16;
  const int b     = qbase >> 12;
  const float scale = 0.125f;          // 64^-0.5

  // per-thread async-staging geometry for the K chunk: 32 keys x 128B,
  // 256 lanes x 16B segments; key = tid>>3, seg = tid&7
  const unsigned ks_lds = (unsigned)(uintptr_t)(&Ks[0][0])
                        + (unsigned)((tid >> 3) * 144 + (tid & 7) * 16);

  // Q A-fragments (16 rows x 64 d -> two K=32 fragments), loaded once
  Frag16 aq[2];
  {
    const __bf16* qrow = q + (size_t)(qbase + ln) * DIM + head * HD;
    #pragma unroll
    for (int h2 = 0; h2 < 2; ++h2)
      #pragma unroll
      for (int j = 0; j < 8; ++j)
        aq[h2].u[j] = *(const unsigned int*)(qrow + h2 * 32 + a_koff(j, hf));
  }

  v8f o[4];
  #pragma unroll
  for (int t = 0; t < 4; ++t) o[t] = {0.f,0.f,0.f,0.f,0.f,0.f,0.f,0.f};
  float mrun[8], lrun[8];
  #pragma unroll
  for (int r = 0; r < 8; ++r) { mrun[r] = -3.0e38f; lrun[r] = 0.f; }

  for (int kc0 = 0; kc0 < KV; kc0 += 32) {
    // async-stage K chunk: one b128 per lane from a contiguous 4KB block
    async_b128(ks_lds, (const char*)(kc + ((size_t)b * KV + kc0) * HD) + tid * 16);

    #pragma unroll
    for (int i = 0; i < 8; ++i) {     // stage V chunk transposed (manual)
      int flat = tid * 8 + i;
      int key = flat >> 6, d = flat & 63;
      Vt[d][key] = vc[((size_t)b * KV + kc0 + key) * HD + d];
    }
    wait_async0();
    __syncthreads();

    // scores S = Q K^T : 16 q x 32 keys (two 16-key N tiles, K-dim = 64)
    v8f s[2];
    #pragma unroll
    for (int nt = 0; nt < 2; ++nt) {
      v8f z = {0.f,0.f,0.f,0.f,0.f,0.f,0.f,0.f};
      const int key = nt * 16 + ln;
      #pragma unroll
      for (int h2 = 0; h2 < 2; ++h2) {
        Frag16 bk;
        #pragma unroll
        for (int j = 0; j < 8; ++j)
          bk.u[j] = *(const unsigned int*)&Ks[key][h2 * 32 + hf * 16 + 2 * j];
        z = wmma_bf16(aq[h2].v, bk.v, z);
      }
      s[nt] = z;
    }

    // online softmax per row (row = r + 8*hf; 16-lane half holds one row set)
    float alpha[8];
    #pragma unroll
    for (int r = 0; r < 8; ++r) {
      float s0 = s[0][r] * scale, s1 = s[1][r] * scale;
      float mx = fmaxf(s0, s1);
      #pragma unroll
      for (int off = 1; off < 16; off <<= 1)
        mx = fmaxf(mx, __shfl_xor(mx, off, 32));
      const float mnew = fmaxf(mrun[r], mx);
      const float al = __expf(mrun[r] - mnew);
      const float p0 = __expf(s0 - mnew);
      const float p1 = __expf(s1 - mnew);
      float rs = p0 + p1;
      #pragma unroll
      for (int off = 1; off < 16; off <<= 1)
        rs += __shfl_xor(rs, off, 32);
      lrun[r] = lrun[r] * al + rs;
      mrun[r] = mnew;
      alpha[r] = al;
      Ps[wave][r + hf * 8][ln]      = (__bf16)p0;   // C-layout -> LDS
      Ps[wave][r + hf * 8][16 + ln] = (__bf16)p1;
    }
    #pragma unroll
    for (int t = 0; t < 4; ++t)
      #pragma unroll
      for (int r = 0; r < 8; ++r)
        o[t][r] *= alpha[r];

    // wave-private LDS round trip: lanes are lockstep, only DS completion needed
    asm volatile("s_wait_dscnt 0" ::: "memory");

    Frag16 ap;                        // P as A-operand (16 x 32)
    #pragma unroll
    for (int j = 0; j < 8; ++j)
      ap.u[j] = *(const unsigned int*)&Ps[wave][ln][a_koff(j, hf)];

    #pragma unroll
    for (int t = 0; t < 4; ++t) {     // O += P * V
      Frag16 bv;
      const int d = t * 16 + ln;
      #pragma unroll
      for (int j = 0; j < 8; ++j)
        bv.u[j] = *(const unsigned int*)&Vt[d][hf * 16 + 2 * j];
      o[t] = wmma_bf16(ap.v, bv.v, o[t]);
    }
    __syncthreads();
  }

  #pragma unroll
  for (int t = 0; t < 4; ++t)
    #pragma unroll
    for (int r = 0; r < 8; ++r) {
      const int row = qbase + r + hf * 8;
      const float val = o[t][r] / lrun[r];
      ao[(size_t)row * DIM + head * HD + t * 16 + ln] = (__bf16)val;
    }
}

// ===========================================================================
// Kernel 4: output projection  A[32768,256] bf16 x out_w[256,256] + out_b
// grid (256, 4). A slab async-copied to LDS; B converted fp32->bf16 manually.
// ===========================================================================
__global__ __launch_bounds__(256) void outproj_kernel(
    const __bf16* __restrict__ a_in,
    const float* __restrict__ w, const float* __restrict__ bias,
    float* __restrict__ out)
{
  __shared__ __bf16 As[128][40];   // pitch 80B (5x16B)
  __shared__ __bf16 Bs[64][40];

  const int tid  = threadIdx.x;
  const int lane = tid & 31;
  const int wave = tid >> 5;
  const int hf   = lane >> 4;
  const int ln   = lane & 15;
  const int m0   = blockIdx.x * 128;
  const int nb   = blockIdx.y;
  const float* wsrc = w + nb * 64;

  const unsigned as_lds0 = (unsigned)(uintptr_t)(&As[0][0]);

  v8f acc[4];
  #pragma unroll
  for (int t = 0; t < 4; ++t) acc[t] = {0.f,0.f,0.f,0.f,0.f,0.f,0.f,0.f};

  for (int k0 = 0; k0 < DIM; k0 += 32) {
    // async-stage A: 128 rows x 64B; 512 b128 segments, 2 per lane
    #pragma unroll
    for (int ss = 0; ss < 2; ++ss) {
      const int sg = tid + ss * 256;
      const int r = sg >> 2, s4 = sg & 3;
      async_b128(as_lds0 + (unsigned)(r * 80 + s4 * 16),
                 (const char*)(a_in + (size_t)(m0 + r) * DIM + k0) + s4 * 16);
    }
    #pragma unroll
    for (int i = 0; i < 8; ++i) {     // B: 32x64 fp32 -> bf16, transposed
      int flat = tid * 8 + i;
      int kk = flat >> 6, n = flat & 63;
      Bs[n][kk] = (__bf16)wsrc[(size_t)(k0 + kk) * DIM + n];
    }
    wait_async0();
    __syncthreads();

    Frag16 a;
    const int arow = wave * 16 + ln;
    #pragma unroll
    for (int j = 0; j < 8; ++j)
      a.u[j] = *(const unsigned int*)&As[arow][a_koff(j, hf)];

    #pragma unroll
    for (int t = 0; t < 4; ++t) {
      Frag16 bfr;
      const int n = t * 16 + ln;
      #pragma unroll
      for (int j = 0; j < 8; ++j)
        bfr.u[j] = *(const unsigned int*)&Bs[n][hf * 16 + 2 * j];
      acc[t] = wmma_bf16(a.v, bfr.v, acc[t]);
    }
    __syncthreads();
  }

  #pragma unroll
  for (int t = 0; t < 4; ++t) {
    const int n = t * 16 + ln;
    const float bv = bias[nb * 64 + n];
    #pragma unroll
    for (int r = 0; r < 8; ++r) {
      const int m = m0 + wave * 16 + r + hf * 8;
      out[(size_t)m * DIM + nb * 64 + n] = acc[t][r] + bv;
    }
  }
}

// ===========================================================================
extern "C" void kernel_launch(void* const* d_in, const int* in_sizes, int n_in,
                              void* d_out, int out_size, void* d_ws, size_t ws_size,
                              hipStream_t stream) {
  const float* x       = (const float*)d_in[0];
  const float* q_w     = (const float*)d_in[1];
  const float* q_b     = (const float*)d_in[2];
  const float* k_w     = (const float*)d_in[3];
  const float* k_b     = (const float*)d_in[4];
  const float* v_w     = (const float*)d_in[5];
  const float* v_b     = (const float*)d_in[6];
  const float* kconv_w = (const float*)d_in[7];
  const float* kconv_b = (const float*)d_in[8];
  const float* vconv_w = (const float*)d_in[9];
  const float* vconv_b = (const float*)d_in[10];
  const float* out_w   = (const float*)d_in[11];
  const float* out_b   = (const float*)d_in[12];
  float* out = (float*)d_out;

  // workspace carve (all sizes multiples of 256B)
  char* p = (char*)d_ws;
  __bf16* q_bf  = (__bf16*)p; p += (size_t)M_ * DIM * 2;   // 16 MB
  float*  k_f   = (float*)p;  p += (size_t)M_ * HD * 4;    //  8 MB
  float*  v_f   = (float*)p;  p += (size_t)M_ * HD * 4;    //  8 MB
  __bf16* kc_bf = (__bf16*)p; p += (size_t)B_ * KV * HD * 2;
  __bf16* vc_bf = (__bf16*)p; p += (size_t)B_ * KV * HD * 2;
  __bf16* at_bf = (__bf16*)p; p += (size_t)M_ * DIM * 2;   // 16 MB
  if ((size_t)(p - (char*)d_ws) > ws_size) return;         // insufficient scratch

  proj_kernel<<<dim3(M_ / 128, 6), 256, 0, stream>>>(
      x, q_w, q_b, k_w, k_b, v_w, v_b, q_bf, k_f, v_f);

  dwconv_kernel<<<dim3((B_ * KV * HD) / 256, 2), 256, 0, stream>>>(
      k_f, kconv_w, kconv_b, v_f, vconv_w, vconv_b, kc_bf, vc_bf);

  attn_kernel<<<dim3(M_ / 32), 256, 0, stream>>>(q_bf, kc_bf, vc_bf, at_bf);

  outproj_kernel<<<dim3(M_ / 128, 4), 256, 0, stream>>>(at_bf, out_w, out_b, out);
}
